// BipartiteSAGE_75935021793804
// MI455X (gfx1250) — compile-verified, matrix operand order checked by reading, not run on hardware
//
#include <hip/hip_runtime.h>
#include <hip/hip_bf16.h>
#include <stdint.h>

// ---------------- CDNA5 WMMA / TDM types ----------------
typedef __attribute__((ext_vector_type(16))) __bf16 v16bf;
typedef __attribute__((ext_vector_type(2)))  __bf16 bf16x2;
typedef __attribute__((ext_vector_type(8)))  float  v8f;
typedef __attribute__((ext_vector_type(4)))  unsigned int u32x4;
typedef __attribute__((ext_vector_type(8)))  int i32x8;
typedef __attribute__((ext_vector_type(4)))  int i32x4;

#if __has_builtin(__builtin_amdgcn_tensor_load_to_lds) && __has_builtin(__builtin_amdgcn_s_wait_tensorcnt)
#define TDM_OK 1
#else
#define TDM_OK 0
#endif

#define NSRC   10000
#define NDST   10000
#define NTOT   20000
#define INSRC  512
#define INDST  256
#define HIDW   512
#define OUTW   256
#define BN_EPS 1e-5f

// ---------------- bf16 helpers (hardware cvt via __bf16 casts) ----------------
__device__ __forceinline__ unsigned pk2(float lo, float hi) {
    union { bf16x2 v; unsigned u; } cv;
    cv.v.x = (__bf16)lo;
    cv.v.y = (__bf16)hi;
    return cv.u;
}
__device__ __forceinline__ unsigned short f2bf(float f) {
    union { __bf16 v; unsigned short u; } cv;
    cv.v = (__bf16)f;
    return cv.u;
}
__device__ __forceinline__ float bf2f(unsigned bits16) {
    return __uint_as_float(bits16 << 16);
}

// Fragment: 16 bf16 elements per lane = 8 VGPRs
union Frag {
    uint4 q[2];
    v16bf v;
};

// ISA 16-bit A/B layout (16x32): lane<16 -> row=lane, K {k0..k0+7} then {k0+16..k0+23};
// lane>=16 -> row=lane-16, K {k0+8..} / {k0+24..}.
__device__ __forceinline__ void load_frag_f32(Frag& f, const float* __restrict__ base,
                                              int row, int K, int k0half) {
    const float4* p = (const float4*)(base + (size_t)row * K + k0half);
    float4 a = p[0], b = p[1];
    float4 c = p[4], d = p[5];
    f.q[0] = make_uint4(pk2(a.x, a.y), pk2(a.z, a.w), pk2(b.x, b.y), pk2(b.z, b.w));
    f.q[1] = make_uint4(pk2(c.x, c.y), pk2(c.z, c.w), pk2(d.x, d.y), pk2(d.z, d.w));
}
// Fragment from an LDS panel ([rows][32 k] bf16, row stride 32 elems)
__device__ __forceinline__ void frag_from_lds(Frag& f, const unsigned short* buf,
                                              int rowLocal, int half) {
    const uint4* p = (const uint4*)(buf + rowLocal * 32 + half * 8);
    f.q[0] = p[0];
    f.q[1] = p[2];
}

// ---- Tensor Data Mover: DMA one 32(k) x tileRows bf16 panel into LDS ----
// D# per CDNA5 ISA: group0 = {count/lds_addr/global_addr/type}, group1 = dims/strides.
// OOB rows (beyond availRows) are zero-filled by the TDM.
__device__ __forceinline__ void tdm_issue(const unsigned short* g, int K, int availRows,
                                          int tileRows, int rowBlk0, int k0,
                                          unsigned short* buf) {
#if TDM_OK
    unsigned long long ga = (unsigned long long)(uintptr_t)(g + (size_t)rowBlk0 * K + k0);
    unsigned ldsa = (unsigned)(uintptr_t)buf;          // low 32 bits = LDS byte address
    unsigned td0 = (unsigned)(K - k0);                 // tensor_dim0 (elements)
    unsigned td1 = (unsigned)(availRows - rowBlk0);    // tensor_dim1 (valid rows)
    u32x4 g0;
    g0[0] = 1u;                                        // count=1, user descriptor
    g0[1] = ldsa;                                      // lds_addr
    g0[2] = (unsigned)(ga & 0xFFFFFFFFu);              // global_addr[31:0]
    g0[3] = (unsigned)((ga >> 32) & 0x01FFFFFFu) | (2u << 30); // addr[56:32] | type=2
    i32x8 g1;
    g1[0] = (int)(1u << 16);                           // workgroup_mask=0, data_size=1 (2B)
    g1[1] = (int)((td0 & 0xFFFFu) << 16);              // tensor_dim0[15:0] @ bits 63:48
    g1[2] = (int)((td0 >> 16) | ((td1 & 0xFFFFu) << 16)); // dim0 hi | dim1 lo
    g1[3] = (int)((td1 >> 16) | (32u << 16));          // dim1 hi | tile_dim0=32
    g1[4] = (int)(unsigned)tileRows;                   // tile_dim1, tile_dim2=0
    g1[5] = (int)(unsigned)K;                          // tensor_dim0_stride (elements)
    g1[6] = 0;
    g1[7] = 0;
    i32x4 z4 = {0, 0, 0, 0};
#if defined(__clang_major__) && (__clang_major__ >= 23)
    i32x8 z8 = {0, 0, 0, 0, 0, 0, 0, 0};
    __builtin_amdgcn_tensor_load_to_lds(g0, g1, z4, z4, z8, 0);
#else
    __builtin_amdgcn_tensor_load_to_lds(g0, g1, z4, z4, 0);
#endif
#else
    (void)g; (void)K; (void)availRows; (void)tileRows; (void)rowBlk0; (void)k0; (void)buf;
#endif
}

// Fallback: cooperative synchronous panel copies (global_load_b128 + ds_store_b128)
__device__ __forceinline__ void panel_copy_sync_b(const unsigned short* W, int K,
                                                  int nblk0, int k0, unsigned short* buf) {
    int c = threadIdx.x;                 // 512 x 16B chunks, 2 per thread
#pragma unroll
    for (int i = 0; i < 2; ++i, c += 256) {
        int row = c >> 2, ch = c & 3;
        *(uint4*)(buf + row * 32 + ch * 8) =
            *(const uint4*)(W + (size_t)(nblk0 + row) * K + k0 + ch * 8);
    }
}
__device__ __forceinline__ void panel_copy_sync_a(const unsigned short* A, int K, int M,
                                                  int mblk0, int k0, unsigned short* buf) {
    int c = threadIdx.x;                 // 256 x 16B chunks, 1 per thread
    int row = c >> 2, ch = c & 3;
    int gr = min(mblk0 + row, M - 1);    // clamp (stores are guarded anyway)
    *(uint4*)(buf + row * 32 + ch * 8) =
        *(const uint4*)(A + (size_t)gr * K + k0 + ch * 8);
}

// ---------------- Dual GEMM: C[M,N] = A1[M,K1]*W1[N,K1]^T + A2[M,K2]*W2[N,K2]^T + bias ----------------
// 256 threads = 8 waves; block tile 64(M) x 128(N); wave tile 32x32 = 2x2 WMMA tiles.
// B panels (and A panels in the bf16 case) staged in LDS via double-buffered TDM DMA.
template <bool AF32, bool OUTBF>
__global__ void __launch_bounds__(256)
gemm_dual_kernel(const void* __restrict__ A1, const unsigned short* __restrict__ W1, int K1,
                 const void* __restrict__ A2, const unsigned short* __restrict__ W2, int K2,
                 const float* __restrict__ bias, void* __restrict__ C, int M, int N) {
    __shared__ unsigned short ldsB[2][128 * 32];   // two 8KB weight panels
    __shared__ unsigned short ldsA[2][64 * 32];    // two 4KB activation panels (bf16-A only)

    const int lane = threadIdx.x & 31;
    const int wave = threadIdx.x >> 5;
    const int r    = lane & 15;
    const int half = lane >> 4;

    const int mblk0 = blockIdx.y * 64;
    const int m0    = mblk0 + (wave >> 2) * 32;
    const int nblk0 = blockIdx.x * 128;
    const int n0    = nblk0 + (wave & 3) * 32;
    const int a0L   = (wave >> 2) * 32 + r;        // LDS-panel-local A rows
    const int a1L   = a0L + 16;
    const int b0L   = (wave & 3) * 32 + r;         // LDS-panel-local B rows
    const int b1L   = b0L + 16;
    const int arow0 = min(m0 + r,      M - 1);     // clamps used by the AF32 path only
    const int arow1 = min(m0 + 16 + r, M - 1);

    v8f c00 = {0.f,0.f,0.f,0.f,0.f,0.f,0.f,0.f};
    v8f c01 = c00, c10 = c00, c11 = c00;

    const int n1 = K1 >> 5, n2 = K2 >> 5;
    const int nsteps = n1 + n2;
    const bool issuer = (threadIdx.x < 32);        // wave 0 drives the TDM

#if TDM_OK
    if (issuer) {
        tdm_issue(W1, K1, N, 128, nblk0, 0, ldsB[0]);
        if constexpr (!AF32)
            tdm_issue((const unsigned short*)A1, K1, M, 64, mblk0, 0, ldsA[0]);
    }
#endif
    for (int s = 0; s < nsteps; ++s) {
        const bool s1 = (s < n1);
        const int k0 = (s1 ? s : s - n1) * 32;
#if TDM_OK
        if (issuer) {
            const int nx = s + 1;
            if (nx < nsteps) {      // prefetch next panels while waiting on current
                const bool x1 = (nx < n1);
                const int kx = (x1 ? nx : nx - n1) * 32;
                tdm_issue(x1 ? W1 : W2, x1 ? K1 : K2, N, 128, nblk0, kx, ldsB[nx & 1]);
                if constexpr (!AF32) {
                    tdm_issue(x1 ? (const unsigned short*)A1 : (const unsigned short*)A2,
                              x1 ? K1 : K2, M, 64, mblk0, kx, ldsA[nx & 1]);
                    __builtin_amdgcn_s_wait_tensorcnt(2);
                } else {
                    __builtin_amdgcn_s_wait_tensorcnt(1);
                }
            } else {
                __builtin_amdgcn_s_wait_tensorcnt(0);
            }
        }
        __syncthreads();            // panels for step s visible to all waves
#else
        (void)issuer;
        __syncthreads();
        panel_copy_sync_b(s1 ? W1 : W2, s1 ? K1 : K2, nblk0, k0, ldsB[s & 1]);
        if constexpr (!AF32)
            panel_copy_sync_a(s1 ? (const unsigned short*)A1 : (const unsigned short*)A2,
                              s1 ? K1 : K2, M, mblk0, k0, ldsA[s & 1]);
        __syncthreads();
#endif
        Frag a0, a1, b0, b1;
        if constexpr (AF32) {
            load_frag_f32(a0, (const float*)A1, arow0, K1, k0 + half * 8);
            load_frag_f32(a1, (const float*)A1, arow1, K1, k0 + half * 8);
        } else {
            frag_from_lds(a0, ldsA[s & 1], a0L, half);
            frag_from_lds(a1, ldsA[s & 1], a1L, half);
        }
        frag_from_lds(b0, ldsB[s & 1], b0L, half);
        frag_from_lds(b1, ldsB[s & 1], b1L, half);
        c00 = __builtin_amdgcn_wmma_f32_16x16x32_bf16(false, a0.v, false, b0.v, (short)0, c00, false, false);
        c01 = __builtin_amdgcn_wmma_f32_16x16x32_bf16(false, a0.v, false, b1.v, (short)0, c01, false, false);
        c10 = __builtin_amdgcn_wmma_f32_16x16x32_bf16(false, a1.v, false, b0.v, (short)0, c10, false, false);
        c11 = __builtin_amdgcn_wmma_f32_16x16x32_bf16(false, a1.v, false, b1.v, (short)0, c11, false, false);
        __syncthreads();            // protect panel buffers before refill
    }

    // C layout: VGPR v, lanes 0-15 -> M=m0+v, N=n; lanes 16-31 -> M=m0+8+v, N=n
    const float bv0 = bias[n0 + r];
    const float bv1 = bias[n0 + 16 + r];
    const int mbase = m0 + half * 8;
#pragma unroll
    for (int v = 0; v < 8; ++v) {
        int ma = mbase + v;
        int mb = ma + 16;
        if (OUTBF) {
            unsigned short* Cb = (unsigned short*)C;
            if (ma < M) {
                Cb[(size_t)ma * N + n0 + r]      = f2bf(c00[v] + bv0);
                Cb[(size_t)ma * N + n0 + 16 + r] = f2bf(c01[v] + bv1);
            }
            if (mb < M) {
                Cb[(size_t)mb * N + n0 + r]      = f2bf(c10[v] + bv0);
                Cb[(size_t)mb * N + n0 + 16 + r] = f2bf(c11[v] + bv1);
            }
        } else {
            float* Cf = (float*)C;
            if (ma < M) {
                Cf[(size_t)ma * N + n0 + r]      = c00[v] + bv0;
                Cf[(size_t)ma * N + n0 + 16 + r] = c01[v] + bv1;
            }
            if (mb < M) {
                Cf[(size_t)mb * N + n0 + r]      = c10[v] + bv0;
                Cf[(size_t)mb * N + n0 + 16 + r] = c11[v] + bv1;
            }
        }
    }
}

// ---------------- elementwise / graph kernels ----------------
__global__ void fill_zero_kernel(float* __restrict__ p, int n) {
    int i = blockIdx.x * blockDim.x + threadIdx.x;
    if (i < n) p[i] = 0.f;
}

__global__ void cvt_bf16_kernel(const float* __restrict__ in, unsigned short* __restrict__ out, int n4) {
    int i = blockIdx.x * blockDim.x + threadIdx.x;
    if (i >= n4) return;
    float4 v = ((const float4*)in)[i];
    uint2 pk;
    pk.x = pk2(v.x, v.y);
    pk.y = pk2(v.z, v.w);
    ((uint2*)out)[i] = pk;
}

__global__ void degree_kernel(const long long* __restrict__ dst, float* __restrict__ cnt, int E) {
    int i = blockIdx.x * blockDim.x + threadIdx.x;
    if (i < E) atomicAdd(&cnt[(int)dst[i]], 1.0f);
}

__global__ void inv_deg_kernel(float* __restrict__ cnt, int n) {
    int i = blockIdx.x * blockDim.x + threadIdx.x;
    if (i < n) cnt[i] = 1.0f / fmaxf(cnt[i], 1.0f);
}

// 64 threads per edge, 8 channels each (uint4 bf16 gather + f32 atomic scatter-add)
__global__ void scatter_add_kernel(const unsigned short* __restrict__ xb,
                                   const long long* __restrict__ src,
                                   const long long* __restrict__ dst,
                                   float* __restrict__ agg, int E) {
    int idx = blockIdx.x * blockDim.x + threadIdx.x;
    int e = idx >> 6;
    if (e >= E) return;
    int c8 = (idx & 63) << 3;
    int s = (int)src[e];
    int d = (int)dst[e];
    uint4 q = *(const uint4*)(xb + (size_t)s * HIDW + c8);
    float* o = agg + (size_t)d * HIDW + c8;
    atomicAdd(o + 0, bf2f(q.x & 0xFFFFu)); atomicAdd(o + 1, bf2f(q.x >> 16));
    atomicAdd(o + 2, bf2f(q.y & 0xFFFFu)); atomicAdd(o + 3, bf2f(q.y >> 16));
    atomicAdd(o + 4, bf2f(q.z & 0xFFFFu)); atomicAdd(o + 5, bf2f(q.z >> 16));
    atomicAdd(o + 6, bf2f(q.w & 0xFFFFu)); atomicAdd(o + 7, bf2f(q.w >> 16));
}

__global__ void scale_cvt_kernel(const float* __restrict__ agg, const float* __restrict__ inv,
                                 unsigned short* __restrict__ ob, int rows) {
    int idx = blockIdx.x * blockDim.x + threadIdx.x;
    int total = rows * (HIDW / 4);
    if (idx >= total) return;
    int row = idx / (HIDW / 4);
    float s = inv[row];
    float4 v = ((const float4*)agg)[idx];
    uint2 pk;
    pk.x = pk2(v.x * s, v.y * s);
    pk.y = pk2(v.z * s, v.w * s);
    ((uint2*)ob)[idx] = pk;
}

// per-channel sum / sumsq -> stats[0..511]=sum, stats[512..1023]=sumsq
__global__ void bn_stats_kernel(const float* __restrict__ y, float* __restrict__ stats, int rows) {
    int c0 = threadIdx.x;
    int c1 = threadIdx.x + 256;
    float s0 = 0.f, s1 = 0.f, q0 = 0.f, q1 = 0.f;
    for (int r = blockIdx.x; r < rows; r += gridDim.x) {
        float a = y[(size_t)r * HIDW + c0];
        float b = y[(size_t)r * HIDW + c1];
        s0 += a; q0 += a * a;
        s1 += b; q1 += b * b;
    }
    atomicAdd(&stats[c0], s0);
    atomicAdd(&stats[c1], s1);
    atomicAdd(&stats[512 + c0], q0);
    atomicAdd(&stats[512 + c1], q1);
}

__global__ void bn_relu_kernel(const float* __restrict__ y, const float* __restrict__ stats,
                               const float* __restrict__ gamma, const float* __restrict__ beta,
                               unsigned short* __restrict__ xo, int rows) {
    int idx = blockIdx.x * blockDim.x + threadIdx.x;
    int total = rows * (HIDW / 4);
    if (idx >= total) return;
    float4 v = ((const float4*)y)[idx];
    int c4 = (idx & (HIDW / 4 - 1)) * 4;
    float invN = 1.0f / (float)rows;
    float a[4] = {v.x, v.y, v.z, v.w};
    float o[4];
#pragma unroll
    for (int j = 0; j < 4; ++j) {
        int c = c4 + j;
        float mu  = stats[c] * invN;
        float var = stats[512 + c] * invN - mu * mu;
        float t = gamma[c] * (a[j] - mu) * rsqrtf(var + BN_EPS) + beta[c];
        o[j] = t > 0.f ? t : 0.f;
    }
    uint2 pk;
    pk.x = pk2(o[0], o[1]);
    pk.y = pk2(o[2], o[3]);
    ((uint2*)xo)[idx] = pk;
}

// ---------------- host launch ----------------
extern "C" void kernel_launch(void* const* d_in, const int* in_sizes, int n_in,
                              void* d_out, int out_size, void* d_ws, size_t ws_size,
                              hipStream_t stream) {
    const float* x_src = (const float*)d_in[0];
    const float* x_dst = (const float*)d_in[1];
    const float* W_src = (const float*)d_in[2];
    const float* b_src = (const float*)d_in[3];
    const float* W_dst = (const float*)d_in[4];
    const float* b_dst = (const float*)d_in[5];
    const float* W1l   = (const float*)d_in[6];
    const float* b1    = (const float*)d_in[7];
    const float* W1r   = (const float*)d_in[8];
    const float* W2l   = (const float*)d_in[9];
    const float* b2    = (const float*)d_in[10];
    const float* W2r   = (const float*)d_in[11];
    const float* gamma = (const float*)d_in[12];
    const float* beta  = (const float*)d_in[13];
    const long long* ei = (const long long*)d_in[14];   // int64 per reference
    const int E = in_sizes[14] / 2;
    const long long* src = ei;
    const long long* dst = ei + E;

    // workspace carve-out (256B aligned)
    char* wp = (char*)d_ws;
    auto carve = [&](size_t bytes) -> void* {
        void* p = (void*)wp;
        wp += (bytes + 255) & ~(size_t)255;
        return p;
    };
    unsigned short* h_bf   = (unsigned short*)carve((size_t)NTOT * HIDW * 2);
    float*          agg    = (float*)         carve((size_t)NTOT * HIDW * 4);
    unsigned short* agg_bf = (unsigned short*)carve((size_t)NTOT * HIDW * 2);
    float*          y1     = (float*)         carve((size_t)NTOT * HIDW * 4);
    unsigned short* x1_bf  = (unsigned short*)carve((size_t)NTOT * HIDW * 2);
    float*          cnt    = (float*)         carve((size_t)NTOT * 4);
    float*          stats  = (float*)         carve(1024 * 4);
    unsigned short* Wsrc_b = (unsigned short*)carve((size_t)HIDW * INSRC * 2);
    unsigned short* Wdst_b = (unsigned short*)carve((size_t)HIDW * INDST * 2);
    unsigned short* W1l_b  = (unsigned short*)carve((size_t)HIDW * HIDW * 2);
    unsigned short* W1r_b  = (unsigned short*)carve((size_t)HIDW * HIDW * 2);
    unsigned short* W2l_b  = (unsigned short*)carve((size_t)OUTW * HIDW * 2);
    unsigned short* W2r_b  = (unsigned short*)carve((size_t)OUTW * HIDW * 2);

    const int T = 256;
    auto blk = [](int n, int t) { return (n + t - 1) / t; };

    // 1) weights -> bf16
    cvt_bf16_kernel<<<blk(HIDW * INSRC / 4, T), T, 0, stream>>>(W_src, Wsrc_b, HIDW * INSRC / 4);
    cvt_bf16_kernel<<<blk(HIDW * INDST / 4, T), T, 0, stream>>>(W_dst, Wdst_b, HIDW * INDST / 4);
    cvt_bf16_kernel<<<blk(HIDW * HIDW / 4, T), T, 0, stream>>>(W1l, W1l_b, HIDW * HIDW / 4);
    cvt_bf16_kernel<<<blk(HIDW * HIDW / 4, T), T, 0, stream>>>(W1r, W1r_b, HIDW * HIDW / 4);
    cvt_bf16_kernel<<<blk(OUTW * HIDW / 4, T), T, 0, stream>>>(W2l, W2l_b, OUTW * HIDW / 4);
    cvt_bf16_kernel<<<blk(OUTW * HIDW / 4, T), T, 0, stream>>>(W2r, W2r_b, OUTW * HIDW / 4);

    // 2) input projections -> h (bf16, concat)
    gemm_dual_kernel<true, true><<<dim3(HIDW / 128, blk(NSRC, 64)), T, 0, stream>>>(
        x_src, Wsrc_b, INSRC, nullptr, Wsrc_b, 0, b_src, h_bf, NSRC, HIDW);
    gemm_dual_kernel<true, true><<<dim3(HIDW / 128, blk(NDST, 64)), T, 0, stream>>>(
        x_dst, Wdst_b, INDST, nullptr, Wdst_b, 0, b_dst, h_bf + (size_t)NSRC * HIDW, NDST, HIDW);

    // 3) degrees
    fill_zero_kernel<<<blk(NTOT, T), T, 0, stream>>>(cnt, NTOT);
    degree_kernel<<<blk(E, T), T, 0, stream>>>(dst, cnt, E);
    inv_deg_kernel<<<blk(NTOT, T), T, 0, stream>>>(cnt, NTOT);

    // 4) layer-1 mean aggregation
    fill_zero_kernel<<<blk(NTOT * HIDW, T), T, 0, stream>>>(agg, NTOT * HIDW);
    scatter_add_kernel<<<blk(E * 64, T), T, 0, stream>>>(h_bf, src, dst, agg, E);
    scale_cvt_kernel<<<blk(NTOT * HIDW / 4, T), T, 0, stream>>>(agg, cnt, agg_bf, NTOT);

    // 5) layer-1 dual GEMM: y1 = agg@W1l^T + b1 + h@W1r^T
    gemm_dual_kernel<false, false><<<dim3(HIDW / 128, blk(NTOT, 64)), T, 0, stream>>>(
        agg_bf, W1l_b, HIDW, h_bf, W1r_b, HIDW, b1, y1, NTOT, HIDW);

    // 6) BatchNorm (training stats) + ReLU -> x1 (bf16)
    fill_zero_kernel<<<blk(1024, T), T, 0, stream>>>(stats, 1024);
    bn_stats_kernel<<<256, 256, 0, stream>>>(y1, stats, NTOT);
    bn_relu_kernel<<<blk(NTOT * HIDW / 4, T), T, 0, stream>>>(y1, stats, gamma, beta, x1_bf, NTOT);

    // 7) layer-2 mean aggregation
    fill_zero_kernel<<<blk(NTOT * HIDW, T), T, 0, stream>>>(agg, NTOT * HIDW);
    scatter_add_kernel<<<blk(E * 64, T), T, 0, stream>>>(x1_bf, src, dst, agg, E);
    scale_cvt_kernel<<<blk(NTOT * HIDW / 4, T), T, 0, stream>>>(agg, cnt, agg_bf, NTOT);

    // 8) layer-2 dual GEMM -> d_out (fp32, [20000,256])
    gemm_dual_kernel<false, false><<<dim3(OUTW / 128, blk(NTOT, 64)), T, 0, stream>>>(
        agg_bf, W2l_b, HIDW, x1_bf, W2r_b, HIDW, b2, (float*)d_out, NTOT, OUTW);

    (void)n_in; (void)out_size; (void)ws_size;
}
